// DecoderWithAttention_62594853371963
// MI455X (gfx1250) — matrix operand order repaired and checked by reading, not compiled, not run on hardware
//
#include <hip/hip_runtime.h>
#include <cstdint>
#include <cstddef>

// ---------------- types & helpers ----------------
typedef __attribute__((ext_vector_type(16))) __bf16 v16bf;
typedef __attribute__((ext_vector_type(2)))  __bf16 v2bf;
typedef __attribute__((ext_vector_type(8)))  float  v8f;

// --- bf16 conversion: prefer native packed cvt; else compiler fptrunc (native on gfx1250) ---
#if __has_builtin(__builtin_amdgcn_cvt_pk_bf16_f32)
__device__ __forceinline__ uint32_t pack2(float lo, float hi) {
    auto p = __builtin_amdgcn_cvt_pk_bf16_f32(lo, hi);
    uint32_t r;
    __builtin_memcpy(&r, &p, 4);
    return r;
}
__device__ __forceinline__ uint16_t f2bf(float f) {
    return (uint16_t)(pack2(f, f) & 0xffffu);
}
#else
__device__ __forceinline__ uint32_t pack2(float lo, float hi) {
    v2bf p;
    p.x = (__bf16)lo;                 // fptrunc -> hw bf16 convert
    p.y = (__bf16)hi;
    uint32_t r;
    __builtin_memcpy(&r, &p, 4);
    return r;
}
__device__ __forceinline__ uint16_t f2bf(float f) {
    __bf16 b = (__bf16)f;
    uint16_t r;
    __builtin_memcpy(&r, &b, 2);
    return r;
}
#endif

__device__ __forceinline__ float bf2f(uint16_t h) {
    return __uint_as_float(((uint32_t)h) << 16);
}

// --- native transcendentals where gfx1250 has them ---
__device__ __forceinline__ float fast_tanh(float x) {
#if __has_builtin(__builtin_amdgcn_tanhf)
    return __builtin_amdgcn_tanhf(x);
#elif __has_builtin(__builtin_amdgcn_tanh_f32)
    return __builtin_amdgcn_tanh_f32(x);
#else
    return tanhf(x);
#endif
}
__device__ __forceinline__ float fast_rcp(float x) {
#if __has_builtin(__builtin_amdgcn_rcpf)
    return __builtin_amdgcn_rcpf(x);
#else
    return 1.0f / x;
#endif
}
__device__ __forceinline__ float sigf(float x) { return fast_rcp(1.0f + __expf(-x)); }

__device__ __forceinline__ v8f wmma_bf16(v16bf a, v16bf b, v8f c) {
    // D = A(16x32 bf16) * B(32x16 bf16) + C(16x16 f32)
    return __builtin_amdgcn_wmma_f32_16x16x32_bf16(false, a, false, b, (short)0, c, false, false);
}

// A tile (16x32 bf16) from an LDS uint16 matrix.
// Layout (ISA 7.12.2): lane m=L%16, half=L/16; element pair j: K = k0 + half*8 + (j&3)*2 + (j>>2)*16
__device__ __forceinline__ v16bf ldsA(const uint16_t* base, int rowStride, int lane) {
    union { v16bf v; uint32_t u[8]; } t;
    int m = lane & 15, half = lane >> 4;
    const uint16_t* p = base + m * rowStride + half * 8;
#pragma unroll
    for (int j = 0; j < 8; ++j) {
        int k = ((j & 3) << 1) + ((j >> 2) << 4);
        t.u[j] = *(const uint32_t*)(p + k);
    }
    return t.v;
}

// B tile from prepacked global buffer: dword index = tile*256 + j*32 + lane (coalesced)
__device__ __forceinline__ v16bf ldB(const uint32_t* __restrict__ pk, int tileIdx, int lane) {
    union { v16bf v; uint32_t u[8]; } t;
    const uint32_t* p = pk + (size_t)tileIdx * 256 + lane;
#pragma unroll
    for (int j = 0; j < 8; ++j) t.u[j] = p[j * 32];
    return t.v;
}

// ---------------- problem constants ----------------
#define NV     96
#define EDIM   64
#define DENC   256
#define HDIM   256
#define ADIM   128
#define NF     10
#define KCONV  5
#define NB     64
#define TENC   1024
#define NSTEP  255
#define BPG    16        // batch elements per workgroup
#define XK     576       // gates GEMM K = 64(emb)+256(ctx)+256(h)

// ---------------- prep kernels ----------------

// Pack B matrix W[N][K] (f32, row-major) into WMMA-ready bf16 dwords.
// B layout: lane N = L%16, half=L/16; VGPR pair j holds K = half*8 + (j&3)*2 + (j>>2)*16 (+0,+1).
__global__ void pack_b_kernel(const float* __restrict__ src, uint32_t* __restrict__ dst,
                              int NT, int KT, int Ksrc, int permute512) {
    int idx = blockIdx.x * blockDim.x + threadIdx.x;
    int total = NT * KT * 256;
    if (idx >= total) return;
    int lane = idx & 31, j = (idx >> 5) & 7, tile = idx >> 8;
    int kt = tile % KT, nt = tile / KT;
    int k = kt * 32 + (lane >> 4) * 8 + ((j & 3) << 1) + ((j >> 2) << 4);
    float f0 = 0.0f, f1 = 0.0f;
    {
        int ks = k;
        if (permute512) ks = (k < 256) ? (k + 256) : (k - 256); // [ctx|h] <- out_w1 cols [h|ctx]
        if (ks < Ksrc) {
            int n0 = nt * 16 + (lane & 15);
            f0 = src[(size_t)n0 * Ksrc + ks];
            f1 = src[(size_t)n0 * Ksrc + ks + 1];
        }
    }
    dst[idx] = pack2(f0, f1);
}

// Combined gates weight: Wg[1024][576] = [W_ih (1024x320) | W_hh (1024x256)]
__global__ void pack_wg_kernel(const float* __restrict__ Wih, const float* __restrict__ Whh,
                               uint32_t* __restrict__ dst) {
    int idx = blockIdx.x * blockDim.x + threadIdx.x;
    int total = 64 * 18 * 256;
    if (idx >= total) return;
    int lane = idx & 31, j = (idx >> 5) & 7, tile = idx >> 8;
    int kt = tile % 18, nt = tile / 18;
    int k = kt * 32 + (lane >> 4) * 8 + ((j & 3) << 1) + ((j >> 2) << 4);
    int n0 = nt * 16 + (lane & 15);
    float f0 = (k < 320) ? Wih[(size_t)n0 * 320 + k] : Whh[(size_t)n0 * 256 + (k - 320)];
    float f1 = (k < 320) ? Wih[(size_t)n0 * 320 + k + 1] : Whh[(size_t)n0 * 256 + (k + 1 - 320)];
    dst[idx] = pack2(f0, f1);
}

__global__ void bias_kernel(const float* __restrict__ bi, const float* __restrict__ bh,
                            float* __restrict__ out) {
    int i = blockIdx.x * blockDim.x + threadIdx.x;
    if (i < 1024) out[i] = bi[i] + bh[i];
}

__global__ void cvt_bf16_kernel(const float* __restrict__ in, uint16_t* __restrict__ out, int n) {
    int i = blockIdx.x * blockDim.x + threadIdx.x;
    if (i < n) out[i] = f2bf(in[i]);
}

// enc_proj = encoder_outputs @ W_enc.T  (65536 x 128, K=256), one 16-row strip per wave.
__global__ __launch_bounds__(256) void encproj_kernel(const float* __restrict__ enc,
                                                      const uint32_t* __restrict__ wenc_pk,
                                                      uint16_t* __restrict__ out) {
    int gw   = (blockIdx.x * blockDim.x + threadIdx.x) >> 5;  // 0..4095, wave-uniform
    int lane = threadIdx.x & 31;
    int row0 = gw * 16;
    int m = lane & 15, half = lane >> 4;

    v16bf A[8];
#pragma unroll
    for (int kt = 0; kt < 8; ++kt) {
        union { v16bf v; uint32_t u[8]; } t;
        const float* src = enc + (size_t)(row0 + m) * 256 + kt * 32 + half * 8;
#pragma unroll
        for (int j = 0; j < 8; ++j) {
            int k = ((j & 3) << 1) + ((j >> 2) << 4);
            t.u[j] = pack2(src[k], src[k + 1]);
        }
        A[kt] = t.v;
    }
#pragma unroll
    for (int nt = 0; nt < 8; ++nt) {
        v8f acc = {0, 0, 0, 0, 0, 0, 0, 0};
#pragma unroll
        for (int kt = 0; kt < 8; ++kt)
            acc = wmma_bf16(A[kt], ldB(wenc_pk, nt * 8 + kt, lane), acc);
        int n = lane & 15, mh = half << 3;
#pragma unroll
        for (int r = 0; r < 8; ++r)
            out[(size_t)(row0 + r + mh) * ADIM + nt * 16 + n] = f2bf(acc[r]);
    }
}

// ---------------- persistent decoder ----------------
__global__ __launch_bounds__(1024) void decoder_kernel(
    const uint16_t* __restrict__ enc_bf,      // [64][1024][256] bf16
    const uint16_t* __restrict__ encproj_bf,  // [64][1024][128] bf16
    const int*      __restrict__ enc_len,     // [64]
    const int*      __restrict__ targets,     // [64][256]
    const float*    __restrict__ emb,         // [96][64]
    const uint32_t* __restrict__ wg_pk,       // gates   NT=64 KT=18
    const float*    __restrict__ biasc,       // [1024]
    const float*    __restrict__ conv_w,      // [10][1][5]
    const uint32_t* __restrict__ wdec_pk,     // NT=8  KT=8
    const uint32_t* __restrict__ wloc_pk,     // NT=8  KT=1 (K padded 10->32)
    const float*    __restrict__ v_w,         // [128]
    const uint32_t* __restrict__ w1_pk,       // NT=16 KT=16 (cols permuted to [ctx|h])
    const float*    __restrict__ out_b1,      // [256]
    const uint32_t* __restrict__ w2_pk,       // NT=6  KT=8
    const float*    __restrict__ out_b2,      // [96]
    float*          __restrict__ out)         // [64][255][96]
{
    __shared__ __align__(16) uint16_t xfull[BPG][XK];   // [emb(64)|ctx(256)|h(256)] bf16
    __shared__ __align__(16) float    gbuf[BPG][TENC];  // gates, then energies/softmax
    __shared__ __align__(16) uint16_t awb[BPG][TENC];   // attention weights (carry) bf16
    __shared__ __align__(16) float    cst[BPG][HDIM];   // LSTM cell state
    __shared__ __align__(16) float    dbuf[BPG][ADIM];  // h @ W_dec.T
    __shared__ __align__(16) uint16_t hidb[BPG][HDIM];  // tanh(out1) bf16
    __shared__           float    red[BPG][64];
    __shared__           float    rowstat[BPG];
    __shared__           float    vbuf[ADIM];
    __shared__           float    cwb[NF][KCONV];

    const int tid  = threadIdx.x;
    const int lane = tid & 31;
    const int wv   = tid >> 5;           // 0..31
    const int b0   = blockIdx.x * BPG;

    // ---- init carries ----
    for (int i = tid; i < BPG * TENC; i += 1024) ((uint16_t*)awb)[i] = f2bf(1.0f / TENC);
    for (int i = tid; i < BPG * HDIM; i += 1024) ((float*)cst)[i] = 0.0f;
    for (int i = tid; i < BPG * 512; i += 1024) {            // ctx + h cols = 0
        int b = i >> 9, c = i & 511;
        xfull[b][64 + c] = 0;
    }
    if (tid < ADIM) vbuf[tid] = v_w[tid];
    if (tid < NF * KCONV) cwb[tid / KCONV][tid % KCONV] = conv_w[tid];
    __syncthreads();

    for (int st = 0; st < NSTEP; ++st) {
        // ---- phase 1: embedding gather into xfull[:, 0:64] ----
        {
            int b = tid >> 6, col = tid & 63;
            int tok = targets[(b0 + b) * 256 + st];
            xfull[b][col] = f2bf(emb[tok * EDIM + col]);
        }
        __syncthreads();

        // ---- phase 2: gates = x_full @ Wg.T  (16 x 1024, K=576) ----
#pragma unroll
        for (int q = 0; q < 2; ++q) {
            int nt = wv * 2 + q;
            v8f acc = {0, 0, 0, 0, 0, 0, 0, 0};
            for (int kt = 0; kt < 18; ++kt)
                acc = wmma_bf16(ldsA(&xfull[0][kt * 32], XK, lane),
                                ldB(wg_pk, nt * 18 + kt, lane), acc);
            int n = lane & 15, mh = (lane >> 4) << 3;
#pragma unroll
            for (int r = 0; r < 8; ++r) gbuf[r + mh][nt * 16 + n] = acc[r];
        }
        __syncthreads();

        // ---- phase 3: LSTM elementwise ----
#pragma unroll
        for (int u = 0; u < 4; ++u) {
            int p = tid + 1024 * u;
            int b = p >> 8, j = p & 255;
            float gi = gbuf[b][j]       + biasc[j];
            float gf = gbuf[b][j + 256] + biasc[j + 256];
            float gg = gbuf[b][j + 512] + biasc[j + 512];
            float go = gbuf[b][j + 768] + biasc[j + 768];
            float c  = sigf(gf) * cst[b][j] + sigf(gi) * fast_tanh(gg);
            float h  = sigf(go) * fast_tanh(c);
            cst[b][j] = c;
            xfull[b][320 + j] = f2bf(h);
        }
        __syncthreads();

        // ---- phase 4: d = h @ W_dec.T  (16 x 128, K=256) ----
        if (wv < 8) {
            int nt = wv;
            v8f acc = {0, 0, 0, 0, 0, 0, 0, 0};
            for (int kt = 0; kt < 8; ++kt)
                acc = wmma_bf16(ldsA(&xfull[0][320 + kt * 32], XK, lane),
                                ldB(wdec_pk, nt * 8 + kt, lane), acc);
            int n = lane & 15, mh = (lane >> 4) << 3;
#pragma unroll
            for (int r = 0; r < 8; ++r) dbuf[r + mh][nt * 16 + n] = acc[r];
        }
        __syncthreads();

        // ---- phase 5: attention energies (fused conv -> WMMA locW -> tanh -> dot v) ----
        for (int it = 0; it < 32; ++it) {
            int tIdx = wv * 32 + it;          // 1024 tiles = 16 b x 64 position-tiles
            int b = tIdx >> 6, pt = tIdx & 63;
            int m = lane & 15, half = lane >> 4;
            int pos = pt * 16 + m;

            // prefetch next tile's enc_proj block (lanes 0..15 cover its 16 rows)
            if (it < 31) {
                int nIdx = tIdx + 1;
                int nb = nIdx >> 6, npt = nIdx & 63;
                const uint16_t* np = encproj_bf +
                    ((size_t)(b0 + nb) * TENC + npt * 16) * ADIM + ((size_t)(lane & 15) * ADIM);
                __builtin_prefetch(np, 0, 3);
            }

            // location conv for this position; half0 -> filters 0..7, half1 -> filters 8,9
            float loc[8];
#pragma unroll
            for (int f = 0; f < 8; ++f) loc[f] = 0.0f;
            if (half == 0) {
#pragma unroll
                for (int f = 0; f < 8; ++f) {
                    float s = 0.0f;
#pragma unroll
                    for (int k = 0; k < KCONV; ++k) {
                        int pp = pos + k - 2;
                        if (pp >= 0 && pp < TENC) s += cwb[f][k] * bf2f(awb[b][pp]);
                    }
                    loc[f] = s;
                }
            } else {
#pragma unroll
                for (int f = 0; f < 2; ++f) {
                    float s = 0.0f;
#pragma unroll
                    for (int k = 0; k < KCONV; ++k) {
                        int pp = pos + k - 2;
                        if (pp >= 0 && pp < TENC) s += cwb[f + 8][k] * bf2f(awb[b][pp]);
                    }
                    loc[f] = s;
                }
            }
            // build A tile (16x32, K = filters padded to 32)
            union { v16bf v; uint32_t u[8]; } ta;
#pragma unroll
            for (int j = 0; j < 8; ++j) ta.u[j] = 0u;
            if (half == 0) {
#pragma unroll
                for (int j = 0; j < 4; ++j) ta.u[j] = pack2(loc[2 * j], loc[2 * j + 1]); // K=0..7
            } else {
                ta.u[0] = pack2(loc[0], loc[1]);                                         // K=8,9
            }

            float esum[8];
#pragma unroll
            for (int r = 0; r < 8; ++r) esum[r] = 0.0f;
            const uint16_t* ep = encproj_bf + ((size_t)(b0 + b) * TENC + pt * 16) * ADIM;
#pragma unroll
            for (int nt = 0; nt < 8; ++nt) {
                v8f acc = {0, 0, 0, 0, 0, 0, 0, 0};
                acc = wmma_bf16(ta.v, ldB(wloc_pk, nt, lane), acc);
                int a = nt * 16 + (lane & 15);
                float dv = dbuf[b][a], vv = vbuf[a];
#pragma unroll
                for (int r = 0; r < 8; ++r) {
                    int mm = r + (half << 3);
                    float x = acc[r] + bf2f(ep[(size_t)mm * ADIM + a]) + dv;
                    esum[r] += vv * fast_tanh(x);
                }
            }
#pragma unroll
            for (int off = 1; off < 16; off <<= 1)
#pragma unroll
                for (int r = 0; r < 8; ++r) esum[r] += __shfl_xor(esum[r], off);
            if ((lane & 15) == 0) {
#pragma unroll
                for (int r = 0; r < 8; ++r) gbuf[b][pt * 16 + r + (half << 3)] = esum[r];
            }
        }
        __syncthreads();

        // ---- phase 6: masked softmax over t_enc per batch row ----
        {
            int b = tid >> 6, j = tid & 63;
            int len = enc_len[b0 + b];
            float mx = -3.0e38f;
            for (int s = 0; s < 16; ++s) {
                int te = j + 64 * s;
                if (te < len) mx = fmaxf(mx, gbuf[b][te]);
            }
            red[b][j] = mx;
        }
        __syncthreads();
        if (tid < BPG) {
            float mx = -3.0e38f;
            for (int j = 0; j < 64; ++j) mx = fmaxf(mx, red[tid][j]);
            rowstat[tid] = mx;
        }
        __syncthreads();
        {
            int b = tid >> 6, j = tid & 63;
            int len = enc_len[b0 + b];
            float mx = rowstat[b], s = 0.0f;
            for (int q = 0; q < 16; ++q) {
                int te = j + 64 * q;
                float e = (te < len) ? __expf(gbuf[b][te] - mx) : 0.0f;
                gbuf[b][te] = e;
                s += e;
            }
            red[b][j] = s;
        }
        __syncthreads();
        if (tid < BPG) {
            float s = 0.0f;
            for (int j = 0; j < 64; ++j) s += red[tid][j];
            rowstat[tid] = fast_rcp(s);
        }
        __syncthreads();
        {
            int b = tid >> 6, j = tid & 63;
            float inv = rowstat[b];
            for (int q = 0; q < 16; ++q) {
                int te = j + 64 * q;
                awb[b][te] = f2bf(gbuf[b][te] * inv);
            }
        }
        __syncthreads();

        // ---- phase 7: ctx = aw @ encoder_outputs  (per-b weighted sum) ----
        {
            int dcol = tid & 255, bq = tid >> 8;   // 4 b's per thread
            float a4[4] = {0, 0, 0, 0};
            for (int te = 0; te < TENC; ++te) {
#pragma unroll
                for (int v4 = 0; v4 < 4; ++v4) {
                    int b = bq * 4 + v4;
                    float a = bf2f(awb[b][te]);
                    a4[v4] += a * bf2f(enc_bf[((size_t)(b0 + b) * TENC + te) * DENC + dcol]);
                }
            }
#pragma unroll
            for (int v4 = 0; v4 < 4; ++v4) xfull[bq * 4 + v4][64 + dcol] = f2bf(a4[v4]);
        }
        __syncthreads();

        // ---- phase 8: hid = tanh([h,ctx] @ out_w1.T + b1)  (A = xfull[:,64:576]) ----
        if (wv < 16) {
            int nt = wv;
            v8f acc = {0, 0, 0, 0, 0, 0, 0, 0};
            for (int kt = 0; kt < 16; ++kt)
                acc = wmma_bf16(ldsA(&xfull[0][64 + kt * 32], XK, lane),
                                ldB(w1_pk, nt * 16 + kt, lane), acc);
            int n = lane & 15, mh = (lane >> 4) << 3;
            float bb = out_b1[nt * 16 + n];
#pragma unroll
            for (int r = 0; r < 8; ++r)
                hidb[r + mh][nt * 16 + n] = f2bf(fast_tanh(acc[r] + bb));
        }
        __syncthreads();

        // ---- phase 9: logits = hid @ out_w2.T + b2 -> global ----
        if (wv < 6) {
            int nt = wv;
            v8f acc = {0, 0, 0, 0, 0, 0, 0, 0};
            for (int kt = 0; kt < 8; ++kt)
                acc = wmma_bf16(ldsA(&hidb[0][kt * 32], HDIM, lane),
                                ldB(w2_pk, nt * 8 + kt, lane), acc);
            int n = lane & 15, mh = (lane >> 4) << 3;
            float bb = out_b2[nt * 16 + n];
#pragma unroll
            for (int r = 0; r < 8; ++r) {
                int b = r + mh;
                out[((size_t)(b0 + b) * NSTEP + st) * NV + nt * 16 + n] = acc[r] + bb;
            }
        }
        __syncthreads();
    }
}

// ---------------- host launch ----------------
extern "C" void kernel_launch(void* const* d_in, const int* in_sizes, int n_in,
                              void* d_out, int out_size, void* d_ws, size_t ws_size,
                              hipStream_t stream) {
    const float* enc      = (const float*)d_in[0];
    const int*   enc_len  = (const int*)  d_in[1];
    const int*   targets  = (const int*)  d_in[2];
    const float* emb      = (const float*)d_in[3];
    const float* W_ih     = (const float*)d_in[4];
    const float* W_hh     = (const float*)d_in[5];
    const float* b_ih     = (const float*)d_in[6];
    const float* b_hh     = (const float*)d_in[7];
    const float* conv_w   = (const float*)d_in[8];
    const float* W_enc    = (const float*)d_in[9];
    const float* W_dec    = (const float*)d_in[10];
    const float* W_loc    = (const float*)d_in[11];
    const float* v_w      = (const float*)d_in[12];
    const float* out_w1   = (const float*)d_in[13];
    const float* out_b1   = (const float*)d_in[14];
    const float* out_w2   = (const float*)d_in[15];
    const float* out_b2   = (const float*)d_in[16];
    float* out = (float*)d_out;

    size_t off = 0;
    auto carve = [&](size_t bytes) -> void* {
        void* p = (char*)d_ws + off;
        off += (bytes + 255) & ~(size_t)255;
        return p;
    };
    uint32_t* wg_pk      = (uint32_t*)carve((size_t)64 * 18 * 256 * 4);
    uint32_t* wenc_pk    = (uint32_t*)carve((size_t)8 * 8 * 256 * 4);
    uint32_t* wdec_pk    = (uint32_t*)carve((size_t)8 * 8 * 256 * 4);
    uint32_t* w1_pk      = (uint32_t*)carve((size_t)16 * 16 * 256 * 4);
    uint32_t* w2_pk      = (uint32_t*)carve((size_t)6 * 8 * 256 * 4);
    uint32_t* wloc_pk    = (uint32_t*)carve((size_t)8 * 1 * 256 * 4);
    float*    biasc      = (float*)   carve(1024 * 4);
    uint16_t* enc_bf     = (uint16_t*)carve((size_t)NB * TENC * DENC * 2);
    uint16_t* encproj_bf = (uint16_t*)carve((size_t)NB * TENC * ADIM * 2);

    // --- weight packing ---
    pack_wg_kernel<<<(64 * 18 * 256 + 255) / 256, 256, 0, stream>>>(W_ih, W_hh, wg_pk);
    pack_b_kernel<<<(8 * 8 * 256 + 255) / 256, 256, 0, stream>>>(W_enc, wenc_pk, 8, 8, 256, 0);
    pack_b_kernel<<<(8 * 8 * 256 + 255) / 256, 256, 0, stream>>>(W_dec, wdec_pk, 8, 8, 256, 0);
    pack_b_kernel<<<(16 * 16 * 256 + 255) / 256, 256, 0, stream>>>(out_w1, w1_pk, 16, 16, 512, 1);
    pack_b_kernel<<<(6 * 8 * 256 + 255) / 256, 256, 0, stream>>>(out_w2, w2_pk, 6, 8, 256, 0);
    pack_b_kernel<<<(8 * 1 * 256 + 255) / 256, 256, 0, stream>>>(W_loc, wloc_pk, 8, 1, 10, 0);
    bias_kernel<<<4, 256, 0, stream>>>(b_ih, b_hh, biasc);

    // --- encoder: bf16 copy + enc_proj GEMM (WMMA) ---
    int nenc = NB * TENC * DENC;
    cvt_bf16_kernel<<<(nenc + 255) / 256, 256, 0, stream>>>(enc, enc_bf, nenc);
    encproj_kernel<<<512, 256, 0, stream>>>(enc, wenc_pk, encproj_bf);

    // --- persistent decoder: 4 workgroups x 16 batch, 255 sequential steps ---
    decoder_kernel<<<4, 1024, 0, stream>>>(enc_bf, encproj_bf, enc_len, targets, emb,
                                           wg_pk, biasc, conv_w, wdec_pk, wloc_pk, v_w,
                                           w1_pk, out_b1, w2_pk, out_b2, out);
}
